// MultiheadAttention_88888643158202
// MI455X (gfx1250) — matrix operand order repaired and checked by reading, not compiled, not run on hardware
//
#include <hip/hip_runtime.h>

// Fused multi-head attention for MI455X (gfx1250).
// bf16 WMMA compute path + Tensor Data Mover (TDM) tile loads into LDS.
// B=2, D=512, N=2048, H=8, head_dim=64, scale = 1/sqrt(512).

#define BB 2
#define DD 512
#define NN 2048
#define HH 8
#define MD 64
#define ATTN_SCALE 0.044194173824159216f  // 1/sqrt(512)

typedef __attribute__((ext_vector_type(16))) __bf16    bf16x16;
typedef __attribute__((ext_vector_type(8)))  __bf16    bf16x8;
typedef __attribute__((ext_vector_type(8)))  float     f32x8;
typedef __attribute__((ext_vector_type(4)))  unsigned  u32x4;
typedef __attribute__((ext_vector_type(8)))  int       i32x8;
typedef __attribute__((ext_vector_type(4)))  int       i32x4;

static __device__ __forceinline__ f32x8 zero8() {
  f32x8 z;
#pragma unroll
  for (int i = 0; i < 8; ++i) z[i] = 0.0f;
  return z;
}

static __device__ __forceinline__ bf16x16 cat16(bf16x8 lo, bf16x8 hi) {
  bf16x16 r;
#pragma unroll
  for (int i = 0; i < 8; ++i) { r[i] = lo[i]; r[i + 8] = hi[i]; }
  return r;
}

// A-matrix (16x32 MxK bf16): lane<16 holds K {0..7,16..23}; lane>=16 holds K {8..15,24..31}.
static __device__ __forceinline__ bf16x16 loadA32(const __bf16* row, int lane) {
  const int o = (lane & 16) ? 8 : 0;
  bf16x8 lo = *(const bf16x8*)(row + o);
  bf16x8 hi = *(const bf16x8*)(row + o + 16);
  return cat16(lo, hi);
}

// B-matrix (32x16 KxN bf16): lane<16 holds K {0..7,16..23} of col lane; lane>=16 holds K {8..15,24..31}.
static __device__ __forceinline__ bf16x16 loadB32(const __bf16* row, int lane) {
  const int o = (lane & 16) ? 8 : 0;
  bf16x8 lo = *(const bf16x8*)(row + o);
  bf16x8 hi = *(const bf16x8*)(row + o + 16);
  return cat16(lo, hi);
}

static __device__ __forceinline__ f32x8 wmma_bf16f32(bf16x16 a, bf16x16 b, f32x8 c) {
  return __builtin_amdgcn_wmma_f32_16x16x32_bf16(
      /*neg_a=*/false, a, /*neg_b=*/false, b,
      /*c_mod=*/(short)0, c, /*reuse_a=*/false, /*reuse_b=*/false);
}

// ---------------------------------------------------------------------------
// Tensor Data Mover: 2D tile load Global -> LDS, bf16 elements (data_size=2B).
// D# layout per CDNA5 ISA (Group0 128b / Group1 256b),
// Group2/3 zero (tensor <= 2D). One issue per calling wave; TENSORcnt tracked.
// ---------------------------------------------------------------------------
static __device__ __forceinline__ void tdm_load_2d_bf16(
    unsigned lds_byte_off, const void* gptr,
    unsigned tile_d0, unsigned tile_d1, unsigned row_stride_elems) {
  const unsigned long long ga = (unsigned long long)(size_t)gptr;
  u32x4 g0;
  g0[0] = 1u;                                        // count=1 (valid), user mode
  g0[1] = lds_byte_off;                              // lds_addr
  g0[2] = (unsigned)(ga & 0xffffffffu);              // global_addr[31:0]
  g0[3] = (unsigned)((ga >> 32) & 0x01ffffffu) | 0x80000000u;  // ga[56:32] | type=2

  i32x8 g1;
  g1[0] = (int)(1u << 16);                           // wg_mask=0, data_size=1 (2B)
  g1[1] = (int)((tile_d0 & 0xffffu) << 16);          // tensor_dim0[15:0] (== tile_d0)
  g1[2] = (int)(((tile_d0 >> 16) & 0xffffu) | ((tile_d1 & 0xffffu) << 16));  // dim0 hi | dim1 lo
  g1[3] = (int)(((tile_d1 >> 16) & 0xffffu) | ((tile_d0 & 0xffffu) << 16));  // dim1 hi | tile_dim0
  g1[4] = (int)(tile_d1 & 0xffffu);                  // tile_dim1 (tile_dim2 = 0)
  g1[5] = (int)row_stride_elems;                     // tensor_dim0_stride[31:0]
  g1[6] = 0;                                         // stride0 hi | dim1_stride lo
  g1[7] = 0;                                         // dim1_stride hi

  const i32x4 z4 = {0, 0, 0, 0};
#if __clang_major__ >= 23
  const i32x8 z8 = {0, 0, 0, 0, 0, 0, 0, 0};
  __builtin_amdgcn_tensor_load_to_lds(g0, g1, z4, z4, z8, 0);
#else
  __builtin_amdgcn_tensor_load_to_lds(g0, g1, z4, z4, 0);
#endif
}

static __device__ __forceinline__ unsigned lds_offset_of(const void* p) {
  return (unsigned)(size_t)p;  // LDS generic address: low 32 bits are the LDS offset
}

// ---------------------------------------------------------------------------
// Projection GEMM: out[b, m, n] = sum_c W[m, c] * X[b, c, n] + bias[m]
// Block: 256 threads = 8 waves arranged 4(M) x 2(N). Block tile 64(M) x 128(N).
// OUT_MODE 0: bf16, head-major TRANSPOSED  [bh][n][d]   (q, k -> TDM-friendly rows)
// OUT_MODE 1: bf16, head-major natural     [bh][d][n]   (v)
// OUT_MODE 2: f32,  [b][m][n]                            (final output proj)
// ---------------------------------------------------------------------------
template <typename TIN, int OUT_MODE>
__global__ __launch_bounds__(256)
void proj_gemm_kernel(const TIN* __restrict__ Xall, const float* __restrict__ W,
                      const float* __restrict__ bias, void* __restrict__ outv) {
  __shared__ __align__(32) __bf16 Ws[64][32];   // W tile: row m, contiguous k
  __shared__ __align__(32) __bf16 Xs[128][32];  // X tile transposed: row n, contiguous k

  const int tid  = threadIdx.x;
  const int lane = tid & 31;
  const int wid  = tid >> 5;
  const int wm   = wid >> 1;  // 0..3
  const int wn   = wid & 1;   // 0..1
  const int n0   = blockIdx.x * 128;
  const int m0   = blockIdx.y * 64;
  const int b    = blockIdx.z;

  const TIN* X = Xall + (size_t)b * DD * NN;

  f32x8 acc[4];
#pragma unroll
  for (int i = 0; i < 4; ++i) acc[i] = zero8();

  for (int kb = 0; kb < DD; kb += 32) {
#pragma unroll
    for (int idx = tid; idx < 64 * 32; idx += 256) {
      const int r = idx >> 5, c = idx & 31;
      Ws[r][c] = (__bf16)W[(size_t)(m0 + r) * DD + kb + c];
    }
#pragma unroll
    for (int idx = tid; idx < 32 * 128; idx += 256) {
      const int r = idx >> 7, n = idx & 127;
      Xs[n][r] = (__bf16)(float)X[(size_t)(kb + r) * NN + n0 + n];
    }
    if (kb + 32 < DD) {
      __builtin_prefetch(&X[(size_t)(kb + 32) * NN + n0 + (tid & 127)], 0, 1);
    }
    __syncthreads();

    const __bf16* arow = &Ws[wm * 16 + (lane & 15)][0];
    const bf16x16 a = loadA32(arow, lane);
#pragma unroll
    for (int nt = 0; nt < 4; ++nt) {
      const __bf16* brow = &Xs[wn * 64 + nt * 16 + (lane & 15)][0];
      acc[nt] = wmma_bf16f32(a, loadB32(brow, lane), acc[nt]);
    }
    __syncthreads();
  }

  // Epilogue: C layout rows r + 8*(lane>=16), col lane%16.
  const int rb = (lane & 16) ? 8 : 0;
#pragma unroll
  for (int nt = 0; nt < 4; ++nt) {
#pragma unroll
    for (int r = 0; r < 8; ++r) {
      const int m = m0 + wm * 16 + r + rb;
      const int n = n0 + wn * 64 + nt * 16 + (lane & 15);
      const float v = acc[nt][r] + bias[m];
      if (OUT_MODE == 0) {
        __bf16* out = (__bf16*)outv;
        const int h = m & (HH - 1);
        const int d = m >> 3;  // channel = d*H + h
        out[((size_t)(b * HH + h) * NN + n) * MD + d] = (__bf16)v;
      } else if (OUT_MODE == 1) {
        __bf16* out = (__bf16*)outv;
        const int h = m & (HH - 1);
        const int d = m >> 3;
        out[((size_t)(b * HH + h) * MD + d) * NN + n] = (__bf16)v;
      } else {
        float* out = (float*)outv;
        out[((size_t)b * DD + m) * NN + n] = v;
      }
    }
  }
}

// ---------------------------------------------------------------------------
// Flash-style attention per head. Block: 128 threads = 4 waves; each block owns
// a 64-query tile, streams keys/values in 64-wide blocks with online softmax.
// q/k in [bh][n][d] bf16 (pre-transposed), v in [bh][d][n] bf16.
// Qs/Ks/Vs tiles are fetched with the Tensor Data Mover (one issuing wave per
// tile, s_wait_tensorcnt + workgroup barrier to publish).
// Output o in channel layout [b][c][n] bf16.
// ---------------------------------------------------------------------------
__global__ __launch_bounds__(128)
void attn_kernel(const __bf16* __restrict__ q_ws, const __bf16* __restrict__ k_ws,
                 const __bf16* __restrict__ v_ws, __bf16* __restrict__ o_ws) {
  __shared__ __align__(32) __bf16 Qs[64][64];      // Q^T tile: [q][d]
  __shared__ __align__(32) __bf16 Ks[64][64];      // K^T tile: [k][d]
  __shared__ __align__(32) __bf16 Vs[64][64];      // V tile:   [d][k]
  __shared__ __align__(32) __bf16 Ps[4][16][64];   // probabilities per wave: [q][k]
  __shared__ float Ss[4][16][64];                  // raw scores per wave
  __shared__ float m_s[64];                        // running row max
  __shared__ float l_s[64];                        // running row sum
  __shared__ float c_s[64];                        // correction factor

  const int tid  = threadIdx.x;
  const int lane = tid & 31;
  const int wid  = tid >> 5;   // wave owns queries [wid*16, wid*16+16)
  const int q0   = blockIdx.x * 64;
  const int bh   = blockIdx.y;
  const int b    = bh >> 3;
  const int h    = bh & 7;

  const __bf16* qh = q_ws + (size_t)bh * MD * NN;  // [n][d]
  const __bf16* kh = k_ws + (size_t)bh * MD * NN;  // [n][d]
  const __bf16* vh = v_ws + (size_t)bh * MD * NN;  // [d][n]

  // Q^T tile via TDM: 64 rows (q) x 64 elems (d), row stride 64.
  if (wid == 2) {
    tdm_load_2d_bf16(lds_offset_of(&Qs[0][0]), qh + (size_t)q0 * MD, 64, 64, MD);
  }
  if (tid < 64) { m_s[tid] = -3.0e38f; l_s[tid] = 0.0f; }

  f32x8 acc[4];  // O^T accumulator: rows = queries, cols = d (4 x 16)
#pragma unroll
  for (int i = 0; i < 4; ++i) acc[i] = zero8();

  for (int kb = 0; kb < NN; kb += 64) {
    __syncthreads();  // previous-iteration consumers done before TDM overwrites
    if (wid == 0) {   // K^T tile: 64 rows (k) x 64 (d), row stride 64
      tdm_load_2d_bf16(lds_offset_of(&Ks[0][0]), kh + (size_t)kb * MD, 64, 64, MD);
    }
    if (wid == 1) {   // V tile: 64 rows (d) x 64 (k), row stride NN
      tdm_load_2d_bf16(lds_offset_of(&Vs[0][0]), vh + kb, 64, 64, NN);
    }
    __builtin_amdgcn_s_wait_tensorcnt(0);  // issuing waves drain; others no-op
    __syncthreads();                        // publish tiles to all waves

    // S = Q^T (16q x 64d) x K (64d x 64k) for this wave's 16 queries.
    const __bf16* aq = &Qs[wid * 16 + (lane & 15)][0];
    const bf16x16 a0 = loadA32(aq, lane);
    const bf16x16 a1 = loadA32(aq + 32, lane);
    const int rb = (lane & 16) ? 8 : 0;
#pragma unroll
    for (int nt = 0; nt < 4; ++nt) {
      const __bf16* bk = &Ks[nt * 16 + (lane & 15)][0];
      f32x8 s = wmma_bf16f32(a0, loadB32(bk, lane), zero8());
      s = wmma_bf16f32(a1, loadB32(bk + 32, lane), s);
#pragma unroll
      for (int r = 0; r < 8; ++r)
        Ss[wid][r + rb][nt * 16 + (lane & 15)] = s[r] * ATTN_SCALE;
    }
    __syncthreads();

    // Online softmax: one thread per query row.
    if (tid < 64) {
      const int q = tid, w = q >> 4, rq = q & 15;
      const float mprev = m_s[q];
      float mx = mprev;
      for (int k = 0; k < 64; ++k) mx = fmaxf(mx, Ss[w][rq][k]);
      float sum = 0.0f;
      for (int k = 0; k < 64; ++k) {
        const float p = __expf(Ss[w][rq][k] - mx);
        Ps[w][rq][k] = (__bf16)p;
        sum += p;
      }
      const float corr = __expf(mprev - mx);
      l_s[q] = l_s[q] * corr + sum;
      m_s[q] = mx;
      c_s[q] = corr;
    }
    __syncthreads();

    // Rescale running O accumulator, then O^T += P (16q x 64k) x V^T (64k x 64d).
    float cr[8];
#pragma unroll
    for (int r = 0; r < 8; ++r) cr[r] = c_s[wid * 16 + r + rb];
#pragma unroll
    for (int dt = 0; dt < 4; ++dt)
#pragma unroll
      for (int r = 0; r < 8; ++r) acc[dt][r] *= cr[r];

    const __bf16* ap = &Ps[wid][lane & 15][0];
    const bf16x16 p0 = loadA32(ap, lane);
    const bf16x16 p1 = loadA32(ap + 32, lane);
#pragma unroll
    for (int dt = 0; dt < 4; ++dt) {
      const __bf16* bv = &Vs[dt * 16 + (lane & 15)][0];
      acc[dt] = wmma_bf16f32(p0, loadB32(bv, lane), acc[dt]);
      acc[dt] = wmma_bf16f32(p1, loadB32(bv + 32, lane), acc[dt]);
    }
  }

  // Normalize and write O in channel layout [b][c = d*H + h][n] as bf16.
  const int rb2 = (lane & 16) ? 8 : 0;
  float inv[8];
#pragma unroll
  for (int r = 0; r < 8; ++r) inv[r] = 1.0f / l_s[wid * 16 + r + rb2];
#pragma unroll
  for (int dt = 0; dt < 4; ++dt)
#pragma unroll
    for (int r = 0; r < 8; ++r) {
      const int d = dt * 16 + (lane & 15);
      const int q = q0 + wid * 16 + r + rb2;
      o_ws[((size_t)b * DD + d * HH + h) * NN + q] = (__bf16)(acc[dt][r] * inv[r]);
    }
}

// ---------------------------------------------------------------------------

extern "C" void kernel_launch(void* const* d_in, const int* in_sizes, int n_in,
                              void* d_out, int out_size, void* d_ws, size_t ws_size,
                              hipStream_t stream) {
  (void)in_sizes; (void)n_in; (void)out_size; (void)ws_size;
  const float* Q  = (const float*)d_in[0];
  const float* Wq = (const float*)d_in[1];
  const float* bq = (const float*)d_in[2];
  const float* Wk = (const float*)d_in[3];
  const float* bk = (const float*)d_in[4];
  const float* Wv = (const float*)d_in[5];
  const float* bv = (const float*)d_in[6];
  const float* Wo = (const float*)d_in[7];
  const float* bo = (const float*)d_in[8];
  float* out = (float*)d_out;

  const size_t perBuf = (size_t)BB * DD * NN;  // elements per bf16 buffer (4 MB)
  __bf16* q_ws = (__bf16*)d_ws;
  __bf16* k_ws = q_ws + perBuf;
  __bf16* v_ws = k_ws + perBuf;
  __bf16* o_ws = v_ws + perBuf;

  const dim3 gproj(NN / 128, DD / 64, BB);
  const dim3 bproj(256);
  proj_gemm_kernel<float, 0><<<gproj, bproj, 0, stream>>>(Q, Wq, bq, (void*)q_ws);
  proj_gemm_kernel<float, 0><<<gproj, bproj, 0, stream>>>(Q, Wk, bk, (void*)k_ws);
  proj_gemm_kernel<float, 1><<<gproj, bproj, 0, stream>>>(Q, Wv, bv, (void*)v_ws);

  const dim3 gattn(NN / 64, BB * HH);
  attn_kernel<<<gattn, dim3(128), 0, stream>>>(q_ws, k_ws, v_ws, o_ws);

  proj_gemm_kernel<__bf16, 2><<<gproj, bproj, 0, stream>>>(o_ws, Wo, bo, (void*)out);
}
